// Attention_68624987455928
// MI455X (gfx1250) — matrix operand order repaired and checked by reading, not compiled, not run on hardware
//
#include <hip/hip_runtime.h>
#include <hip/hip_bf16.h>
#include <math.h>

// Problem sizes (fixed by the reference)
#define BB 8
#define NN 2048
#define DD 512
#define HH 512

// ---------------------------------------------------------------------------
// WMMA types (CDNA5 / gfx1250, wave32)
// ---------------------------------------------------------------------------
typedef __bf16 v16bf __attribute__((ext_vector_type(16)));
typedef float  v8f   __attribute__((ext_vector_type(8)));
typedef int    v4i   __attribute__((ext_vector_type(4)));

__device__ __forceinline__ unsigned short f2bf_bits(float f) {
  union { float f; unsigned int u; } x; x.f = f;
  unsigned int r = x.u + 0x7FFFu + ((x.u >> 16) & 1u);   // round-to-nearest-even
  return (unsigned short)(r >> 16);
}

union Frag {
  v16bf v;
  uint4 u[2];
};

union FragPack {
  v16bf v;
  unsigned short h[16];
};

// LDS tile row stride: 32 data bf16 + 16 pad = 96 bytes (16B-aligned rows,
// spreads banks to cut conflicts on per-lane b128 fragment reads)
#define LDT 48

// ---------------------------------------------------------------------------
// CDNA5 async global->LDS copy (ASYNCcnt) with safe fallback.
// Builtin signature (probed from the Round-2 diagnostic):
//   (int4 AS1* gsrc, int4 AS3* ldsdst, imm offset, imm cpol)
// ---------------------------------------------------------------------------
#if __has_builtin(__builtin_amdgcn_global_load_async_to_lds_b128)
#define HAVE_ASYNC 1
#else
#define HAVE_ASYNC 0
#endif

__device__ __forceinline__ void async_ld_b128(void* lds, const void* g) {
#if HAVE_ASYNC
  __builtin_amdgcn_global_load_async_to_lds_b128(
      (__attribute__((address_space(1))) v4i*)g,
      (__attribute__((address_space(3))) v4i*)lds, /*offset=*/0, /*cpol=*/0);
#else
  *(uint4*)lds = *(const uint4*)g;   // sync fallback (dscnt, covered by barrier)
#endif
}

template <int N>
__device__ __forceinline__ void wait_async() {
#if HAVE_ASYNC
#if __has_builtin(__builtin_amdgcn_s_wait_asynccnt)
  __builtin_amdgcn_s_wait_asynccnt(N);
#else
  asm volatile("s_wait_asynccnt %0" ::"i"(N));
#endif
#endif
}

// A-matrix 16x32 bf16 fragment (ISA 7.12.2): lane L(0..15)/L+16 holds row M=L.
// v0..3 = K[khalf*8 .. +7], v4..7 = K[16+khalf*8 .. +7]   (khalf = lane>>4)
__device__ __forceinline__ void frag_a_from_lds(Frag& f, const unsigned short* rowBase, int khalf) {
  f.u[0] = *(const uint4*)(rowBase + khalf * 8);
  f.u[1] = *(const uint4*)(rowBase + 16 + khalf * 8);
}

// B-matrix 32x16 bf16 fragment: lane holds column N=lane&15; same K order as A:
// v0..3 = K[khalf*8 .. +7], v4..7 = K[16+khalf*8 .. +7]   (khalf = lane>>4)
__device__ __forceinline__ void frag_b_from_lds(Frag& f, const unsigned short* rowBase, int khalf) {
  f.u[0] = *(const uint4*)(rowBase + khalf * 8);
  f.u[1] = *(const uint4*)(rowBase + 16 + khalf * 8);
}

// Stage one 128x32 bf16 tile pair (A and B) via async copies.
// 512 chunks of 8 bf16 per tile; 256 threads -> 2 chunks each per tile.
__device__ __forceinline__ void stage_pair_async(unsigned short* lA, unsigned short* lB,
                                                 const unsigned short* gA, int lda,
                                                 const unsigned short* gB, int ldb,
                                                 int tid) {
#pragma unroll
  for (int p = 0; p < 2; ++p) {
    const int idx = tid + p * 256;
    const int row = idx >> 2, kc = (idx & 3) << 3;
    async_ld_b128(&lA[row * LDT + kc], gA + (size_t)row * lda + kc);
    async_ld_b128(&lB[row * LDT + kc], gB + (size_t)row * ldb + kc);
  }
}

__device__ __forceinline__ void stage_b_async(unsigned short* lB,
                                              const unsigned short* gB, int ldb, int tid) {
#pragma unroll
  for (int p = 0; p < 2; ++p) {
    const int idx = tid + p * 256;
    const int row = idx >> 2, kc = (idx & 3) << 3;
    async_ld_b128(&lB[row * LDT + kc], gB + (size_t)row * ldb + kc);
  }
}

// ---------------------------------------------------------------------------
// x: f32 [B][N][D] -> xbf (row major bf16) and xT (bf16 [B][D][N], transposed)
// ---------------------------------------------------------------------------
__global__ void attn_xconv_kernel(const float* __restrict__ x,
                                  unsigned short* __restrict__ xbf,
                                  unsigned short* __restrict__ xT) {
  __shared__ unsigned short tile[32][33];
  const int b  = blockIdx.z;
  const int n0 = blockIdx.x * 32;
  const int d0 = blockIdx.y * 32;
  const int tx = threadIdx.x, ty = threadIdx.y;

  const float* xb = x + (size_t)b * NN * DD;
  unsigned short v = f2bf_bits(xb[(size_t)(n0 + ty) * DD + (d0 + tx)]);
  xbf[(size_t)b * NN * DD + (size_t)(n0 + ty) * DD + (d0 + tx)] = v;
  tile[ty][tx] = v;
  __syncthreads();
  xT[(size_t)b * DD * NN + (size_t)(d0 + ty) * NN + (n0 + tx)] = tile[tx][ty];
}

// ---------------------------------------------------------------------------
// W: f32 [D][H] -> WT: bf16 [H][D]  (so GEMM B-staging is contiguous rows)
// ---------------------------------------------------------------------------
__global__ void attn_wconv_kernel(const float* __restrict__ W,
                                  unsigned short* __restrict__ WT) {
  __shared__ unsigned short tile[32][33];
  const int d0 = blockIdx.x * 32;
  const int h0 = blockIdx.y * 32;
  const int tx = threadIdx.x, ty = threadIdx.y;
  tile[ty][tx] = f2bf_bits(W[(size_t)(d0 + ty) * HH + (h0 + tx)]);
  __syncthreads();
  WT[(size_t)(h0 + ty) * DD + (d0 + tx)] = tile[tx][ty];
}

// ---------------------------------------------------------------------------
// Generic double-buffered bf16 GEMM body (A row-major [m][k], B rows [n][k]).
// 128x128 block tile, 8 waves as 4(M)x2(N), wave tile 32x64, K step 32.
// ---------------------------------------------------------------------------
#define GEMM_BODY(Abase, lda, Bbase, ldb, KDIM)                                        \
  const int tid = threadIdx.x;                                                         \
  const int wave = tid >> 5, lane = tid & 31;                                          \
  const int wm = wave & 3, wn = wave >> 2;                                             \
  const int r = lane & 15, khalf = lane >> 4;                                          \
  const v8f vzero = {0.f, 0.f, 0.f, 0.f, 0.f, 0.f, 0.f, 0.f};                          \
  v8f acc[2][4];                                                                       \
  for (int mi = 0; mi < 2; ++mi)                                                       \
    for (int ni = 0; ni < 4; ++ni) acc[mi][ni] = vzero;                                \
  stage_pair_async(lA[0], lB[0], (Abase), (lda), (Bbase), (ldb), tid);                 \
  const int KT = (KDIM) / 32;                                                          \
  for (int kt = 0; kt < KT; ++kt) {                                                    \
    const int cur = kt & 1;                                                            \
    if (kt + 1 < KT) {                                                                 \
      stage_pair_async(lA[cur ^ 1], lB[cur ^ 1], (Abase) + (kt + 1) * 32, (lda),       \
                       (Bbase) + (kt + 1) * 32, (ldb), tid);                           \
      wait_async<4>();                                                                 \
    } else {                                                                           \
      wait_async<0>();                                                                 \
    }                                                                                  \
    __syncthreads();                                                                   \
    Frag a[2], bfr[4];                                                                 \
    for (int mi = 0; mi < 2; ++mi)                                                     \
      frag_a_from_lds(a[mi], &lA[cur][(wm * 32 + mi * 16 + r) * LDT], khalf);          \
    for (int ni = 0; ni < 4; ++ni)                                                     \
      frag_b_from_lds(bfr[ni], &lB[cur][(wn * 64 + ni * 16 + r) * LDT], khalf);        \
    for (int mi = 0; mi < 2; ++mi)                                                     \
      for (int ni = 0; ni < 4; ++ni)                                                   \
        acc[mi][ni] = __builtin_amdgcn_wmma_f32_16x16x32_bf16(                         \
            false, a[mi].v, false, bfr[ni].v, (short)0, acc[mi][ni], false, false);    \
    __syncthreads();                                                                   \
  }                                                                                    \
  const int rb = khalf << 3; /* C/D layout: lanes 16-31 hold M=8..15 */

// ---------------------------------------------------------------------------
// Projection GEMM: Out[m][h] = bf16( sum_d xbf[m][d]*WT[h][d] + bias[h] )
// ---------------------------------------------------------------------------
__global__ void __launch_bounds__(256)
attn_proj_kernel(const unsigned short* __restrict__ Abf,
                 const unsigned short* __restrict__ WT,
                 const float* __restrict__ bias,
                 unsigned short* __restrict__ Out) {
  __shared__ unsigned short lA[2][128 * LDT];
  __shared__ unsigned short lB[2][128 * LDT];
  const int m0 = blockIdx.x * 128;
  const int n0 = blockIdx.y * 128;

  GEMM_BODY(Abf + (size_t)m0 * DD, DD, WT + (size_t)n0 * DD, DD, DD)

  for (int mi = 0; mi < 2; ++mi)
    for (int ni = 0; ni < 4; ++ni) {
      const int col = n0 + wn * 64 + ni * 16 + r;
      const int row = m0 + wm * 32 + mi * 16 + rb;
      const float bv = bias[col];
      for (int g = 0; g < 8; ++g)
        Out[(size_t)(row + g) * HH + col] = f2bf_bits(acc[mi][ni][g] + bv);
    }
}

// ---------------------------------------------------------------------------
// Energy GEMM: E[b][i][j] = sum_h Q[b][i][h] * K[b][j][h]   (f32 out)
// ---------------------------------------------------------------------------
__global__ void __launch_bounds__(256)
attn_energy_kernel(const unsigned short* __restrict__ Q,
                   const unsigned short* __restrict__ Kb16,
                   float* __restrict__ E) {
  __shared__ unsigned short lA[2][128 * LDT];
  __shared__ unsigned short lB[2][128 * LDT];
  const int b = blockIdx.z;
  const unsigned short* Qb = Q    + (size_t)b * NN * HH;
  const unsigned short* Kb = Kb16 + (size_t)b * NN * HH;
  float* Eb = E + (size_t)b * NN * NN;
  const int m0 = blockIdx.x * 128;   // i
  const int n0 = blockIdx.y * 128;   // j

  GEMM_BODY(Qb + (size_t)m0 * HH, HH, Kb + (size_t)n0 * HH, HH, HH)

  for (int mi = 0; mi < 2; ++mi)
    for (int ni = 0; ni < 4; ++ni) {
      const int col = n0 + wn * 64 + ni * 16 + r;
      const int row = m0 + wm * 32 + mi * 16 + rb;
      for (int g = 0; g < 8; ++g)
        Eb[(size_t)(row + g) * NN + col] = acc[mi][ni][g];
    }
}

// ---------------------------------------------------------------------------
// Column softmax stats over the QUERY axis i: per (b,j) online max + sum(exp).
// ---------------------------------------------------------------------------
__global__ void attn_colstats_kernel(const float* __restrict__ E,
                                     float* __restrict__ mcol,
                                     float* __restrict__ inv_s) {
  const int col = blockIdx.x * 256 + threadIdx.x;   // flat b*N + j
  const int b = col >> 11;                          // N == 2048
  const int j = col & (NN - 1);
  const float* base = E + (size_t)b * NN * NN + j;

  float mx = base[0];
  float s  = 1.0f;
  for (int i = 1; i < NN; ++i) {
    const float v = base[(size_t)i * NN];
    const float nm = fmaxf(mx, v);
    s = s * __expf(mx - nm) + __expf(v - nm);
    mx = nm;
  }
  mcol[col]  = mx;
  inv_s[col] = 1.0f / s;
}

// ---------------------------------------------------------------------------
// Output GEMM: Out[b][i][d] = sum_j exp(E[i][j]-m_j)*inv_s_j * x[b][j][d]
// A (P) fragments built in registers from L2-resident E: float4 loads +
// v_exp_f32 + bf16 pack (TRANS VALU co-executes with XDL WMMA).
// B = xT rows, async double-buffered in LDS. f32 output to d_out.
// ---------------------------------------------------------------------------
__global__ void __launch_bounds__(256)
attn_out_kernel(const float* __restrict__ E,
                const float* __restrict__ mcol,
                const float* __restrict__ inv_s,
                const unsigned short* __restrict__ xT,
                float* __restrict__ Out) {
  __shared__ unsigned short lB[2][128 * LDT];

  const int b = blockIdx.z;
  const float* Eb = E     + (size_t)b * NN * NN;
  const float* mb = mcol  + (size_t)b * NN;
  const float* ib = inv_s + (size_t)b * NN;
  const unsigned short* xTb = xT + (size_t)b * DD * NN;
  float* Ob = Out + (size_t)b * NN * DD;

  const int tid  = threadIdx.x;
  const int m0   = blockIdx.x * 128;   // i
  const int n0   = blockIdx.y * 128;   // d
  const int wave = tid >> 5, lane = tid & 31;
  const int wm = wave & 3, wn = wave >> 2;
  const int r = lane & 15, khalf = lane >> 4;

  const v8f vzero = {0.f, 0.f, 0.f, 0.f, 0.f, 0.f, 0.f, 0.f};
  v8f acc[2][4];
  for (int mi = 0; mi < 2; ++mi)
    for (int ni = 0; ni < 4; ++ni) acc[mi][ni] = vzero;

  const unsigned short* gB = xTb + (size_t)n0 * NN;
  stage_b_async(lB[0], gB, NN, tid);

  const int KT = NN / 32;
  for (int kt = 0; kt < KT; ++kt) {
    const int cur = kt & 1;
    const int k0  = kt * 32;
    if (kt + 1 < KT) {
      stage_b_async(lB[cur ^ 1], gB + (kt + 1) * 32, NN, tid);
      wait_async<2>();
    } else {
      wait_async<0>();
    }
    __syncthreads();

    // Per-column softmax stats for this K block (shared across both mi frags)
    const int jb = k0 + (khalf << 3);           // low j group of this lane
    const float4 mv0 = *(const float4*)(mb + jb);
    const float4 mv1 = *(const float4*)(mb + jb + 4);
    const float4 mv2 = *(const float4*)(mb + jb + 16);
    const float4 mv3 = *(const float4*)(mb + jb + 20);
    const float4 iv0 = *(const float4*)(ib + jb);
    const float4 iv1 = *(const float4*)(ib + jb + 4);
    const float4 iv2 = *(const float4*)(ib + jb + 16);
    const float4 iv3 = *(const float4*)(ib + jb + 20);

    Frag bfr[4];
    for (int ni = 0; ni < 4; ++ni)
      frag_b_from_lds(bfr[ni], &lB[cur][(wn * 64 + ni * 16 + r) * LDT], khalf);

    FragPack a[2];
    for (int mi = 0; mi < 2; ++mi) {
      const float* e = Eb + (size_t)(m0 + wm * 32 + mi * 16 + r) * NN + jb;
      const float4 e0 = *(const float4*)(e);
      const float4 e1 = *(const float4*)(e + 4);
      const float4 e2 = *(const float4*)(e + 16);
      const float4 e3 = *(const float4*)(e + 20);
      a[mi].h[0]  = f2bf_bits(__expf(e0.x - mv0.x) * iv0.x);
      a[mi].h[1]  = f2bf_bits(__expf(e0.y - mv0.y) * iv0.y);
      a[mi].h[2]  = f2bf_bits(__expf(e0.z - mv0.z) * iv0.z);
      a[mi].h[3]  = f2bf_bits(__expf(e0.w - mv0.w) * iv0.w);
      a[mi].h[4]  = f2bf_bits(__expf(e1.x - mv1.x) * iv1.x);
      a[mi].h[5]  = f2bf_bits(__expf(e1.y - mv1.y) * iv1.y);
      a[mi].h[6]  = f2bf_bits(__expf(e1.z - mv1.z) * iv1.z);
      a[mi].h[7]  = f2bf_bits(__expf(e1.w - mv1.w) * iv1.w);
      a[mi].h[8]  = f2bf_bits(__expf(e2.x - mv2.x) * iv2.x);
      a[mi].h[9]  = f2bf_bits(__expf(e2.y - mv2.y) * iv2.y);
      a[mi].h[10] = f2bf_bits(__expf(e2.z - mv2.z) * iv2.z);
      a[mi].h[11] = f2bf_bits(__expf(e2.w - mv2.w) * iv2.w);
      a[mi].h[12] = f2bf_bits(__expf(e3.x - mv3.x) * iv3.x);
      a[mi].h[13] = f2bf_bits(__expf(e3.y - mv3.y) * iv3.y);
      a[mi].h[14] = f2bf_bits(__expf(e3.z - mv3.z) * iv3.z);
      a[mi].h[15] = f2bf_bits(__expf(e3.w - mv3.w) * iv3.w);
    }

    for (int mi = 0; mi < 2; ++mi)
      for (int ni = 0; ni < 4; ++ni)
        acc[mi][ni] = __builtin_amdgcn_wmma_f32_16x16x32_bf16(
            false, a[mi].v, false, bfr[ni].v, (short)0, acc[mi][ni], false, false);
    __syncthreads();
  }

  const int rb = khalf << 3;
  for (int mi = 0; mi < 2; ++mi)
    for (int ni = 0; ni < 4; ++ni) {
      const int col = n0 + wn * 64 + ni * 16 + r;
      const int row = m0 + wm * 32 + mi * 16 + rb;
      for (int g = 0; g < 8; ++g)
        Ob[(size_t)(row + g) * DD + col] = acc[mi][ni][g];
    }
}

// ---------------------------------------------------------------------------
// Host launcher
// ---------------------------------------------------------------------------
extern "C" void kernel_launch(void* const* d_in, const int* in_sizes, int n_in,
                              void* d_out, int out_size, void* d_ws, size_t ws_size,
                              hipStream_t stream) {
  const float* x  = (const float*)d_in[0];
  const float* Wq = (const float*)d_in[1];
  const float* bq = (const float*)d_in[2];
  const float* Wk = (const float*)d_in[3];
  const float* bk = (const float*)d_in[4];
  float* out = (float*)d_out;

  char* ws = (char*)d_ws;
  size_t off = 0;
  auto wsalloc = [&](size_t bytes) -> void* {
    void* p = ws + off;
    off = (off + bytes + 255) & ~(size_t)255;
    return p;
  };

  unsigned short* xbf = (unsigned short*)wsalloc((size_t)BB * NN * DD * 2);
  unsigned short* xT  = (unsigned short*)wsalloc((size_t)BB * NN * DD * 2);
  unsigned short* WqT = (unsigned short*)wsalloc((size_t)DD * HH * 2);
  unsigned short* WkT = (unsigned short*)wsalloc((size_t)DD * HH * 2);
  unsigned short* Qbf = (unsigned short*)wsalloc((size_t)BB * NN * HH * 2);
  unsigned short* Kbf = (unsigned short*)wsalloc((size_t)BB * NN * HH * 2);
  float* E     = (float*)wsalloc((size_t)BB * NN * NN * 4);
  float* mcol  = (float*)wsalloc((size_t)BB * NN * 4);
  float* inv_s = (float*)wsalloc((size_t)BB * NN * 4);

  attn_xconv_kernel<<<dim3(NN / 32, DD / 32, BB), dim3(32, 32), 0, stream>>>(x, xbf, xT);
  attn_wconv_kernel<<<dim3(DD / 32, HH / 32), dim3(32, 32), 0, stream>>>(Wq, WqT);
  attn_wconv_kernel<<<dim3(DD / 32, HH / 32), dim3(32, 32), 0, stream>>>(Wk, WkT);
  attn_proj_kernel<<<dim3((BB * NN) / 128, HH / 128), 256, 0, stream>>>(xbf, WqT, bq, Qbf);
  attn_proj_kernel<<<dim3((BB * NN) / 128, HH / 128), 256, 0, stream>>>(xbf, WkT, bk, Kbf);
  attn_energy_kernel<<<dim3(NN / 128, NN / 128, BB), 256, 0, stream>>>(Qbf, Kbf, E);
  attn_colstats_kernel<<<dim3((BB * NN) / 256), 256, 0, stream>>>(E, mcol, inv_s);
  attn_out_kernel<<<dim3(NN / 128, DD / 128, BB), 256, 0, stream>>>(E, mcol, inv_s, xT, out);
}
